// TransformerDecoder_10153302688476
// MI455X (gfx1250) — compile-verified, matrix operand order checked by reading, not run on hardware
//
#include <hip/hip_runtime.h>
#include <hip/hip_bf16.h>

// ---------------------------------------------------------------------------
// MoE transformer decoder for gfx1250 (MI455X), bf16 WMMA GEMMs + fp32 rest.
// ---------------------------------------------------------------------------

typedef __attribute__((ext_vector_type(16))) __bf16 bfvec16;
typedef __attribute__((ext_vector_type(8)))  float  v8f;

#define TOK    2048      // B*S tokens
#define BATCH  2
#define SEQ    1024
#define DMODEL 1024
#define DKV    256       // NKV*HEAD
#define HFF    2048
#define NEXP   8
#define NHEAD  16
#define NKVH   4
#define HD     64
#define RMS_EPS 1e-6f

// Native conversion: clang lowers float->__bf16 to the hardware cvt on
// gfx1250 (RNE), replacing the 3-VALU software rounding sequence.
__device__ __forceinline__ __bf16 f2bf(float f) { return (__bf16)f; }

// ---------------------------------------------------------------------------
// WMMA GEMM:  C[M,N] = A[M,K] * B[N,K]^T + bias[N]
//   MODE 0: C = val
//   MODE 1: C = val + resid
//   MODE 2: C += rowscale[m*NEXP] * val     (MoE expert accumulate)
// Block: 256 threads = 8 waves; block tile 128(M) x 128(N); K-step 32.
// Wave tile 32x64 = 2x4 v_wmma_f32_16x16x32_bf16 per K-step.
// Double-buffered LDS (ping-pong) with register-staged global fetch so the
// HBM stream for tile k+1 overlaps WMMA on tile k; one barrier per K-step.
// M, N, K must be multiples of 128/128/32 (true for all call sites).
// ---------------------------------------------------------------------------
#define BM 128
#define BN 128
#define BK 32
#define LDSS 40   // padded LDS row stride (bf16 elems): 80B, kills bank conflicts

template <int MODE>
__global__ __launch_bounds__(256) void gemm_bf16_wmma(
    const float* __restrict__ A, const float* __restrict__ B,
    const float* __restrict__ bias, const float* __restrict__ resid,
    const float* __restrict__ rowscale, float* __restrict__ C,
    int N, int K)
{
  __shared__ __bf16 sA[2][BM][LDSS];
  __shared__ __bf16 sB[2][BN][LDSS];

  const int tid   = threadIdx.x;
  const int lane  = tid & 31;
  const int wave  = tid >> 5;
  const int waveM = wave >> 1;        // 0..3 -> M rows [waveM*32, +32)
  const int waveN = wave & 1;         // 0..1 -> N cols [waveN*64, +64)
  const int r     = lane & 15;        // row within 16
  const int hk    = lane >> 4;        // K-half select (0/1)
  const int blockN = blockIdx.x * BN;
  const int blockM = blockIdx.y * BM;

  v8f acc[2][4];
  const v8f vzero = {0.f, 0.f, 0.f, 0.f, 0.f, 0.f, 0.f, 0.f};
#pragma unroll
  for (int i = 0; i < 2; ++i)
#pragma unroll
    for (int j = 0; j < 4; ++j) acc[i][j] = vzero;

  // register staging for the global->LDS pipeline (A and B tiles are both
  // 128x32 = 4096 fp32 -> 16 elems = 4 x float4 per thread each)
  float4 ra[4], rb[4];

  auto fetch = [&](int k0) {
#pragma unroll
    for (int c = 0; c < 4; ++c) {
      const int ei = tid * 16 + c * 4;
      const int row = ei >> 5, col = ei & 31;
      ra[c] = *(const float4*)(A + (size_t)(blockM + row) * K + k0 + col);
      rb[c] = *(const float4*)(B + (size_t)(blockN + row) * K + k0 + col);
    }
  };
  auto stage = [&](int buf) {
#pragma unroll
    for (int c = 0; c < 4; ++c) {
      const int ei = tid * 16 + c * 4;
      const int row = ei >> 5, col = ei & 31;
      sA[buf][row][col + 0] = f2bf(ra[c].x);
      sA[buf][row][col + 1] = f2bf(ra[c].y);
      sA[buf][row][col + 2] = f2bf(ra[c].z);
      sA[buf][row][col + 3] = f2bf(ra[c].w);
      sB[buf][row][col + 0] = f2bf(rb[c].x);
      sB[buf][row][col + 1] = f2bf(rb[c].y);
      sB[buf][row][col + 2] = f2bf(rb[c].z);
      sB[buf][row][col + 3] = f2bf(rb[c].w);
    }
  };
  auto compute = [&](int buf) {
    // CDNA5 16-bit A/B fragment layout: lane(half=hk,row=r):
    //   elems 0..7 = K[hk*8 .. +7], elems 8..15 = K[16+hk*8 .. +7]
    bfvec16 af[2], bfr[4];
#pragma unroll
    for (int i = 0; i < 2; ++i) {
      const __bf16* p0 = &sA[buf][waveM * 32 + i * 16 + r][hk * 8];
      const __bf16* p1 = &sA[buf][waveM * 32 + i * 16 + r][16 + hk * 8];
#pragma unroll
      for (int t = 0; t < 8; ++t) { af[i][t] = p0[t]; af[i][8 + t] = p1[t]; }
    }
#pragma unroll
    for (int j = 0; j < 4; ++j) {
      const __bf16* p0 = &sB[buf][waveN * 64 + j * 16 + r][hk * 8];
      const __bf16* p1 = &sB[buf][waveN * 64 + j * 16 + r][16 + hk * 8];
#pragma unroll
      for (int t = 0; t < 8; ++t) { bfr[j][t] = p0[t]; bfr[j][8 + t] = p1[t]; }
    }
#pragma unroll
    for (int i = 0; i < 2; ++i)
#pragma unroll
      for (int j = 0; j < 4; ++j)
        acc[i][j] = __builtin_amdgcn_wmma_f32_16x16x32_bf16(
            false, af[i], false, bfr[j], (short)0, acc[i][j], false, false);
  };

  const int KT = K / BK;
  fetch(0);
  stage(0);
  __syncthreads();

  for (int kt = 0; kt < KT; ++kt) {
    const int cur = kt & 1;
    if (kt + 1 < KT) {
      fetch((kt + 1) * BK);                       // HBM stream overlaps WMMA
      if (kt + 2 < KT) {
        const int row = tid >> 1, col = (tid & 1) * 16;
        __builtin_prefetch(A + (size_t)(blockM + row) * K + (kt + 2) * BK + col, 0, 1);
        __builtin_prefetch(B + (size_t)(blockN + row) * K + (kt + 2) * BK + col, 0, 1);
      }
    }
    compute(cur);
    if (kt + 1 < KT) stage(cur ^ 1);
    __syncthreads();
  }

  // ---- epilogue: C/D layout -> lane(half,col): vgpr t holds M = half*8 + t ----
#pragma unroll
  for (int i = 0; i < 2; ++i) {
#pragma unroll
    for (int j = 0; j < 4; ++j) {
      const int n  = blockN + waveN * 64 + j * 16 + r;
      const int mb = blockM + waveM * 32 + i * 16 + hk * 8;
      const float bv = bias[n];
#pragma unroll
      for (int t = 0; t < 8; ++t) {
        const int m = mb + t;
        const size_t idx = (size_t)m * N + n;
        const float val = acc[i][j][t] + bv;
        if (MODE == 0) C[idx] = val;
        if (MODE == 1) C[idx] = val + resid[idx];
        if (MODE == 2) C[idx] += rowscale[(size_t)m * NEXP] * val;
      }
    }
  }
}

// ---------------------------------------------------------------------------
// Global L2-norm (reference sums x*x over the ENTIRE tensor) — deterministic
// two-stage reduction, then elementwise scale by w[d]/(sqrt(S)+eps).
// ---------------------------------------------------------------------------
__global__ __launch_bounds__(256) void sumsq_partial(const float* __restrict__ x,
                                                     float* __restrict__ part) {
  __shared__ float red[256];
  const size_t base = (size_t)blockIdx.x * 1024;
  float a = 0.f;
  for (int i = threadIdx.x; i < 1024; i += 256) { float v = x[base + i]; a += v * v; }
  red[threadIdx.x] = a;
  __syncthreads();
  for (int o = 128; o > 0; o >>= 1) {
    if (threadIdx.x < o) red[threadIdx.x] += red[threadIdx.x + o];
    __syncthreads();
  }
  if (threadIdx.x == 0) part[blockIdx.x] = red[0];
}

__global__ __launch_bounds__(256) void sumsq_final(const float* __restrict__ part,
                                                   float* __restrict__ out) {
  __shared__ float red[256];
  float a = 0.f;
  for (int i = threadIdx.x; i < 2048; i += 256) a += part[i];
  red[threadIdx.x] = a;
  __syncthreads();
  for (int o = 128; o > 0; o >>= 1) {
    if (threadIdx.x < o) red[threadIdx.x] += red[threadIdx.x + o];
    __syncthreads();
  }
  if (threadIdx.x == 0) out[0] = red[0];
}

__global__ __launch_bounds__(256) void rms_scale(const float* __restrict__ x,
                                                 const float* __restrict__ w,
                                                 const float* __restrict__ sumsq,
                                                 float* __restrict__ y, int n) {
  int i = blockIdx.x * 256 + threadIdx.x;
  if (i >= n) return;
  const float inv = 1.0f / (sqrtf(sumsq[0]) + RMS_EPS);
  y[i] = w[i & (DMODEL - 1)] * x[i] * inv;
}

// ---------------------------------------------------------------------------
// RoPE (interleaved pairs), in place.  layout: [(b*S+s), nh*64]
// ---------------------------------------------------------------------------
__global__ __launch_bounds__(256) void rope_kernel(float* __restrict__ q, int nh, int total) {
  int idx = blockIdx.x * 256 + threadIdx.x;
  if (idx >= total) return;
  const int i  = idx & 31;          // pair index 0..31
  const int hs = idx >> 5;          // = (b*S+s)*nh + h
  const int s  = (hs / nh) % SEQ;   // position
  const float inv = powf(10000.0f, -(float)i / 32.0f);
  const float f = (float)s * inv;
  const float c = cosf(f), sn = sinf(f);
  const size_t base = (size_t)hs * 64 + 2 * i;
  const float x0 = q[base], x1 = q[base + 1];
  q[base]     = x0 * c - x1 * sn;
  q[base + 1] = x0 * sn + x1 * c;
}

// ---------------------------------------------------------------------------
// Causal GQA attention, fp32.  grid = (SEQ, NHEAD, BATCH), 128 threads.
// kv head = h % NKVH (reference uses jnp.tile, not repeat).
// ---------------------------------------------------------------------------
__global__ __launch_bounds__(128) void attn_kernel(const float* __restrict__ Q,
                                                   const float* __restrict__ K,
                                                   const float* __restrict__ V,
                                                   float* __restrict__ O) {
  const int qpos = blockIdx.x, h = blockIdx.y, b = blockIdx.z;
  const int kvh = h % NKVH;
  const int tid = threadIdx.x;
  const int L = qpos + 1;

  __shared__ float qv[HD];
  __shared__ float sc[SEQ];
  __shared__ float red[128];

  const float* qp = Q + (size_t)(b * SEQ + qpos) * DMODEL + h * HD;
  if (tid < HD) qv[tid] = qp[tid];
  __syncthreads();

  float lmax = -3.402823466e38f;
  for (int k = tid; k < L; k += 128) {
    const float* kp = K + (size_t)(b * SEQ + k) * DKV + kvh * HD;
    float d = 0.f;
#pragma unroll
    for (int j = 0; j < HD; ++j) d += qv[j] * kp[j];
    d *= 0.125f;                       // 1/sqrt(64)
    sc[k] = d;
    lmax = fmaxf(lmax, d);
  }
  red[tid] = lmax;
  __syncthreads();
  for (int o = 64; o > 0; o >>= 1) {
    if (tid < o) red[tid] = fmaxf(red[tid], red[tid + o]);
    __syncthreads();
  }
  const float m = red[0];
  __syncthreads();

  float lsum = 0.f;
  for (int k = tid; k < L; k += 128) { float p = expf(sc[k] - m); sc[k] = p; lsum += p; }
  red[tid] = lsum;
  __syncthreads();
  for (int o = 64; o > 0; o >>= 1) {
    if (tid < o) red[tid] += red[tid + o];
    __syncthreads();
  }
  const float ssum = red[0];
  __syncthreads();

  const int d = tid & (HD - 1);
  const int half = tid >> 6;
  float acc = 0.f;
  for (int k = half; k < L; k += 2)
    acc += sc[k] * V[(size_t)(b * SEQ + k) * DKV + kvh * HD + d];
  red[tid] = acc;
  __syncthreads();
  if (half == 0)
    O[(size_t)(b * SEQ + qpos) * DMODEL + h * HD + d] = (red[tid] + red[tid + 64]) / ssum;
}

// ---------------------------------------------------------------------------
// Gate: logits = x @ gate^T (8), top-2 softmax -> wf[TOK][8] (zeros elsewhere).
// ---------------------------------------------------------------------------
__global__ __launch_bounds__(256) void gate_topk(const float* __restrict__ X,
                                                 const float* __restrict__ G,
                                                 float* __restrict__ wf) {
  int t = blockIdx.x * 256 + threadIdx.x;
  if (t >= TOK) return;
  const float* x = X + (size_t)t * DMODEL;
  float l[NEXP];
#pragma unroll
  for (int e = 0; e < NEXP; ++e) l[e] = 0.f;
  for (int d = 0; d < DMODEL; ++d) {
    const float xv = x[d];
#pragma unroll
    for (int e = 0; e < NEXP; ++e) l[e] += xv * G[e * DMODEL + d];
  }
  int i0 = 0;
  for (int e = 1; e < NEXP; ++e) if (l[e] > l[i0]) i0 = e;
  int i1 = -1;
  for (int e = 0; e < NEXP; ++e) {
    if (e == i0) continue;
    if (i1 < 0 || l[e] > l[i1]) i1 = e;
  }
  const float e1 = expf(l[i1] - l[i0]);
  const float inv = 1.0f / (1.0f + e1);
  float* w = wf + (size_t)t * NEXP;
#pragma unroll
  for (int e = 0; e < NEXP; ++e) w[e] = 0.f;
  w[i0] = inv;
  w[i1] = e1 * inv;
}

// ---------------------------------------------------------------------------
// elementwise helpers
// ---------------------------------------------------------------------------
__global__ __launch_bounds__(256) void silu_mul(float* __restrict__ g0,
                                                const float* __restrict__ g1, int n) {
  int i = blockIdx.x * 256 + threadIdx.x;
  if (i >= n) return;
  const float z = g1[i];
  g0[i] = g0[i] * (z / (1.0f + expf(-z)));
}

__global__ __launch_bounds__(256) void fill_zero(float* __restrict__ p, int n) {
  int i = blockIdx.x * 256 + threadIdx.x;
  if (i < n) p[i] = 0.f;
}

__global__ __launch_bounds__(256) void add_kernel(const float* __restrict__ a,
                                                  const float* __restrict__ b,
                                                  float* __restrict__ c, int n) {
  int i = blockIdx.x * 256 + threadIdx.x;
  if (i < n) c[i] = a[i] + b[i];
}

// ---------------------------------------------------------------------------
// host orchestration
// ---------------------------------------------------------------------------
extern "C" void kernel_launch(void* const* d_in, const int* in_sizes, int n_in,
                              void* d_out, int out_size, void* d_ws, size_t ws_size,
                              hipStream_t stream) {
  (void)in_sizes; (void)n_in; (void)out_size; (void)ws_size;

  const float* x_in = (const float*)d_in[0];
  float* ws = (float*)d_ws;
  size_t o = 0;
  float* b_norm = ws + o; o += (size_t)TOK * DMODEL;
  float* b_q    = ws + o; o += (size_t)TOK * DMODEL;
  float* b_k    = ws + o; o += (size_t)TOK * DKV;
  float* b_v    = ws + o; o += (size_t)TOK * DKV;
  float* b_att  = ws + o; o += (size_t)TOK * DMODEL;
  float* b_h1   = ws + o; o += (size_t)TOK * DMODEL;
  float* b_g0   = ws + o; o += (size_t)TOK * HFF;
  float* b_g1   = ws + o; o += (size_t)TOK * HFF;
  float* b_moe  = ws + o; o += (size_t)TOK * DMODEL;
  float* b_x    = ws + o; o += (size_t)TOK * DMODEL;
  float* b_wf   = ws + o; o += (size_t)TOK * NEXP;
  float* b_part = ws + o; o += 2048;
  float* b_scal = ws + o; o += 4;

  const int NTOT = TOK * DMODEL;           // 2,097,152
  const float* cur_x = x_in;

  for (int l = 0; l < 2; ++l) {
    const int base = 1 + l * 59;           // per-layer: 11 top-level + 8*6 expert arrays
    const float* att_norm = (const float*)d_in[base + 0];
    const float* qw = (const float*)d_in[base + 1];
    const float* qb = (const float*)d_in[base + 2];
    const float* kw = (const float*)d_in[base + 3];
    const float* kb = (const float*)d_in[base + 4];
    const float* vw = (const float*)d_in[base + 5];
    const float* vb = (const float*)d_in[base + 6];
    const float* ow = (const float*)d_in[base + 7];
    const float* ob = (const float*)d_in[base + 8];
    const float* ffn_norm = (const float*)d_in[base + 9];
    const float* gate = (const float*)d_in[base + 10];

    // ---- attention block ----
    sumsq_partial<<<2048, 256, 0, stream>>>(cur_x, b_part);
    sumsq_final<<<1, 256, 0, stream>>>(b_part, b_scal);
    rms_scale<<<NTOT / 256, 256, 0, stream>>>(cur_x, att_norm, b_scal, b_norm, NTOT);

    gemm_bf16_wmma<0><<<dim3(DMODEL / BN, TOK / BM), 256, 0, stream>>>(
        b_norm, qw, qb, nullptr, nullptr, b_q, DMODEL, DMODEL);
    gemm_bf16_wmma<0><<<dim3(DKV / BN, TOK / BM), 256, 0, stream>>>(
        b_norm, kw, kb, nullptr, nullptr, b_k, DKV, DMODEL);
    gemm_bf16_wmma<0><<<dim3(DKV / BN, TOK / BM), 256, 0, stream>>>(
        b_norm, vw, vb, nullptr, nullptr, b_v, DKV, DMODEL);

    const int ropeQ = BATCH * SEQ * NHEAD * 32;
    const int ropeK = BATCH * SEQ * NKVH * 32;
    rope_kernel<<<(ropeQ + 255) / 256, 256, 0, stream>>>(b_q, NHEAD, ropeQ);
    rope_kernel<<<(ropeK + 255) / 256, 256, 0, stream>>>(b_k, NKVH, ropeK);

    attn_kernel<<<dim3(SEQ, NHEAD, BATCH), 128, 0, stream>>>(b_q, b_k, b_v, b_att);

    // h1 = cur_x + att @ ow^T + ob   (residual fused into GEMM epilogue)
    gemm_bf16_wmma<1><<<dim3(DMODEL / BN, TOK / BM), 256, 0, stream>>>(
        b_att, ow, ob, cur_x, nullptr, b_h1, DMODEL, DMODEL);

    // ---- MoE block ----
    sumsq_partial<<<2048, 256, 0, stream>>>(b_h1, b_part);
    sumsq_final<<<1, 256, 0, stream>>>(b_part, b_scal);
    rms_scale<<<NTOT / 256, 256, 0, stream>>>(b_h1, ffn_norm, b_scal, b_norm, NTOT);

    gate_topk<<<TOK / 256, 256, 0, stream>>>(b_norm, gate, b_wf);
    fill_zero<<<NTOT / 256, 256, 0, stream>>>(b_moe, NTOT);

    for (int e = 0; e < NEXP; ++e) {
      const int eb = base + 11 + e * 6;
      const float* w0w = (const float*)d_in[eb + 0];
      const float* w0b = (const float*)d_in[eb + 1];
      const float* w1w = (const float*)d_in[eb + 2];
      const float* w1b = (const float*)d_in[eb + 3];
      const float* w2w = (const float*)d_in[eb + 4];
      const float* w2b = (const float*)d_in[eb + 5];

      gemm_bf16_wmma<0><<<dim3(HFF / BN, TOK / BM), 256, 0, stream>>>(
          b_norm, w0w, w0b, nullptr, nullptr, b_g0, HFF, DMODEL);
      gemm_bf16_wmma<0><<<dim3(HFF / BN, TOK / BM), 256, 0, stream>>>(
          b_norm, w1w, w1b, nullptr, nullptr, b_g1, HFF, DMODEL);
      silu_mul<<<(TOK * HFF) / 256, 256, 0, stream>>>(b_g0, b_g1, TOK * HFF);
      // moe += wf[:,e] * (g0 @ w2^T + b2)   (routing weight fused into epilogue)
      gemm_bf16_wmma<2><<<dim3(DMODEL / BN, TOK / BM), 256, 0, stream>>>(
          b_g0, w2w, w2b, nullptr, b_wf + e, b_moe, DMODEL, HFF);
    }

    float* nx = (l == 1) ? (float*)d_out : b_x;
    add_kernel<<<NTOT / 256, 256, 0, stream>>>(b_h1, b_moe, nx, NTOT);
    cur_x = nx;
  }
}
